// Merge_MixtralSparseMoeBlock_14559939134022
// MI455X (gfx1250) — compile-verified, matrix operand order checked by reading, not run on hardware
//
#include <hip/hip_runtime.h>
#include <hip/hip_bf16.h>

// ---------------------------------------------------------------------------
// Mixtral MoE block (all experts aliased) for MI455X / gfx1250, wave32.
// f32 inputs converted to bf16 while staging into LDS; all GEMMs on
// v_wmma_f32_16x16x32_bf16 (f32 accumulate).  bf16 intermediates (h, t*) are
// staged into LDS with global_load_async_to_lds_b128 (ASYNCcnt) when
// available.  __launch_bounds__(256,2) gives each wave ~512 VGPRs so the
// 8-16 accumulator octets never spill to scratch.
//
// Workspace layout (~134 MB):
//   [0]      scale : 8192 f32
//   [64KB]   t1    : 8192 x 448 bf16   (x @ v1^T, cols 398..447 zero)
//   +7.0MB   t3    : 8192 x 448 bf16   (x @ v3^T)
//   +7.0MB   t2    : 8192 x 448 bf16   (h @ v2^T)
//   +7.0MB   h     : 8192 x 7168 bf16  (silu(gate) * up)
// ---------------------------------------------------------------------------

#define USE_ASYNC_LDS 1   // gfx1250 GLOBAL_LOAD_ASYNC_TO_LDS_B128 for bf16 tiles

typedef __attribute__((ext_vector_type(16))) __bf16 v16bf;
typedef __attribute__((ext_vector_type(8)))  __bf16 v8bf;
typedef __attribute__((ext_vector_type(4)))  __bf16 v4bf;
typedef __attribute__((ext_vector_type(8)))  float  v8f;

#define BM 128
#define BN 128
#define BK 64
#define PITCH 72            // bf16 elems per LDS row: 144B stride, 16B aligned
#define TLE (128 * PITCH)   // LDS elems per 128x64 tile buffer
#define BIGROW (1LL << 40)

#define NTOK  8192LL
#define HDIM  2048LL
#define IDIM  7168LL
#define RPAD  448LL         // 398 padded up to multiple of BK
#define RREAL 398LL

// ---------------- WMMA helpers -------------------------------------------

static __device__ __forceinline__ v8f wmma_bf16(v16bf a, v16bf b, v8f c) {
  return __builtin_amdgcn_wmma_f32_16x16x32_bf16(false, a, false, b,
                                                 (short)0, c, false, false);
}

// A-frag (16x32) / B-frag (32x16 from [out,K] layout): identical LDS pattern.
// lanes 0-15: row = base+lane16, K = kb+0..7 (VGPR0-3) and kb+16..23 (VGPR4-7)
// lanes16-31: same rows, K offset by +8 (per ISA 16-bit A/B layout tables).
static __device__ __forceinline__ v16bf ld_frag(const __bf16* sm, int rowb,
                                                int kb, int lane16, int hi) {
  const __bf16* p = sm + (rowb + lane16) * PITCH + kb + hi * 8;
  union { v16bf w; v8bf h[2]; } u;
  u.h[0] = *(const v8bf*)(p);
  u.h[1] = *(const v8bf*)(p + 16);
  return u.w;
}

// One BK=64 slab: 2 k-steps x (2 M-frags x 4 N-frags) = 16 WMMAs into acc.
static __device__ __forceinline__ void mma_step(const __bf16* smA,
                                                const __bf16* smB,
                                                v8f acc[][4], int wm, int wn,
                                                int lane16, int hi) {
#pragma unroll
  for (int kk = 0; kk < BK; kk += 32) {
    v16bf a0 = ld_frag(smA, wm * 32 + 0,  kk, lane16, hi);
    v16bf a1 = ld_frag(smA, wm * 32 + 16, kk, lane16, hi);
#pragma unroll
    for (int nf = 0; nf < 4; ++nf) {
      v16bf b = ld_frag(smB, wn * 64 + nf * 16, kk, lane16, hi);
      acc[0][nf] = wmma_bf16(a0, b, acc[0][nf]);
      acc[1][nf] = wmma_bf16(a1, b, acc[1][nf]);
    }
  }
}

// Barrier that also drains outstanding async-to-LDS transfers of this wave.
static __device__ __forceinline__ void tile_barrier() {
#if USE_ASYNC_LDS
  asm volatile("s_wait_asynccnt 0x0" ::: "memory");
#endif
  __syncthreads();
}

// ---------------- cooperative tile loaders (256 threads) ------------------

// 128x64 f32 tile -> bf16 LDS (vectorized; requires ld % 4 == 0).
// Rows >= rowlim are zero-filled (used for v* matrices with only 398 rows).
static __device__ __forceinline__ void ld_tile_f32v4(__bf16* sm, const float* g,
                                                     long row0, long col0,
                                                     long ld, long rowlim,
                                                     int tid) {
#pragma unroll
  for (int i = 0; i < 8; ++i) {
    int lin = tid + 256 * i;          // 0..2047
    int r   = lin >> 4;               // 0..127
    int c4  = (lin & 15) << 2;        // 0,4,..,60
    float4 f = make_float4(0.f, 0.f, 0.f, 0.f);
    long gr = row0 + r;
    if (gr < rowlim) f = *(const float4*)(g + gr * ld + col0 + c4);
    v4bf h = {(__bf16)f.x, (__bf16)f.y, (__bf16)f.z, (__bf16)f.w};
    *(v4bf*)(sm + r * PITCH + c4) = h;
  }
}

// Scalar variant with K-bound (used for u* matrices: ld = 398, K padded).
static __device__ __forceinline__ void ld_tile_f32s(__bf16* sm, const float* g,
                                                    long row0, long col0,
                                                    long ld, long klim,
                                                    int tid) {
#pragma unroll
  for (int i = 0; i < 8; ++i) {
    int lin = tid + 256 * i;
    int r   = lin >> 4;
    int c4  = (lin & 15) << 2;
    long gr = row0 + r;
    v4bf h;
#pragma unroll
    for (int j = 0; j < 4; ++j) {
      long kc = col0 + c4 + j;
      float f = (kc < klim) ? g[gr * ld + kc] : 0.f;
      h[j] = (__bf16)f;
    }
    *(v4bf*)(sm + r * PITCH + c4) = h;
  }
}

// 128x64 bf16 tile -> LDS.  Uses gfx1250 async L2->LDS DMA when enabled:
// per-lane 16B chunk, LDS dest offset in VDST vgpr, tracked by ASYNCcnt.
static __device__ __forceinline__ void ld_tile_bf16(__bf16* sm, const __bf16* g,
                                                    long row0, long col0,
                                                    long ld, int tid) {
#pragma unroll
  for (int i = 0; i < 4; ++i) {
    int lin = tid + 256 * i;          // 0..1023
    int r   = lin >> 3;               // 0..127
    int c8  = (lin & 7) << 3;         // 0,8,..,56
    const __bf16* gp = g + (row0 + r) * ld + col0 + c8;
    __bf16* lp = sm + r * PITCH + c8;
#if USE_ASYNC_LDS
    unsigned lds_off = (unsigned)(unsigned long long)lp;  // flat low 32 = LDS addr
    asm volatile("global_load_async_to_lds_b128 %0, %1, off"
                 :: "v"(lds_off), "v"(gp)
                 : "memory");
#else
    *(v8bf*)lp = *(const v8bf*)gp;
#endif
  }
}

// overload dispatch for templated A operand
static __device__ __forceinline__ void ld_tileA(__bf16* sm, const float* g,
                                                long row0, long col0, long ld,
                                                int tid) {
  ld_tile_f32v4(sm, g, row0, col0, ld, BIGROW, tid);
}
static __device__ __forceinline__ void ld_tileA(__bf16* sm, const __bf16* g,
                                                long row0, long col0, long ld,
                                                int tid) {
  ld_tile_bf16(sm, g, row0, col0, ld, tid);
}

// ---------------- kernel 1: router logits + per-token scale ---------------

__global__ void __launch_bounds__(256, 2)
k_router(const float* __restrict__ X, const float* __restrict__ GW,
         float* __restrict__ logits, float* __restrict__ scale) {
  __shared__ float sl[8];
  const long n  = blockIdx.x;
  const int tid = threadIdx.x;
  const int wv  = tid >> 5;
  const int ln  = tid & 31;
  const float* xr = X + n * HDIM;
  const float* gr = GW + (long)wv * HDIM;
  float s = 0.f;
  for (int i = ln; i < (int)HDIM; i += 32) s += xr[i] * gr[i];
#pragma unroll
  for (int off = 16; off > 0; off >>= 1) s += __shfl_down(s, off, 32);
  if (ln == 0) sl[wv] = s;
  __syncthreads();
  if (tid == 0) {
    float mx = sl[0];
    for (int e = 1; e < 8; ++e) mx = fmaxf(mx, sl[e]);
    float p[8], Z = 0.f;
    for (int e = 0; e < 8; ++e) { p[e] = __expf(sl[e] - mx); Z += p[e]; }
    float a = -1.f, b = -1.f;
    for (int e = 0; e < 8; ++e) {
      if (p[e] > a) { b = a; a = p[e]; } else if (p[e] > b) { b = p[e]; }
    }
    float pa = a / Z, pb = b / Z, sm2 = pa + pb;
    scale[n] = pa / sm2 + pb / sm2;
  }
  if (tid < 8) logits[n * 8 + tid] = sl[tid];
}

// ---------------- kernel 2/5: low-rank projection  T = A @ V^T ------------
// A: [8192, K] (f32 or bf16), V: [398, K] f32, T: [8192, 448] bf16 (padded 0)

template <typename AT>
__global__ void __launch_bounds__(256, 2)
k_lowrank(const AT* __restrict__ A, const float* __restrict__ V,
          __bf16* __restrict__ T, long K) {
  __shared__ __bf16 smA[TLE];
  __shared__ __bf16 smB[TLE];
  const int tid = threadIdx.x;
  const long m0 = (long)blockIdx.x * BM;
  const long n0 = (long)blockIdx.y * BN;
  const int lane = tid & 31, wv = tid >> 5;
  const int wm = wv & 3, wn = wv >> 2;
  const int lane16 = lane & 15, hi = lane >> 4;
  const v8f z = {0.f, 0.f, 0.f, 0.f, 0.f, 0.f, 0.f, 0.f};
  v8f acc[2][4];
  for (int a = 0; a < 2; ++a)
    for (int b = 0; b < 4; ++b) acc[a][b] = z;

  for (long k0 = 0; k0 < K; k0 += BK) {
    ld_tileA(smA, A, m0, k0, K, tid);
    ld_tile_f32v4(smB, V, n0, k0, K, RREAL, tid);   // rows >= 398 -> zeros
    tile_barrier();
    mma_step(smA, smB, acc, wm, wn, lane16, hi);
    __syncthreads();
  }
#pragma unroll
  for (int mf = 0; mf < 2; ++mf)
#pragma unroll
    for (int nf = 0; nf < 4; ++nf) {
      long col = n0 + wn * 64 + nf * 16 + lane16;
      if (col < RPAD) {
        long rowb = m0 + wm * 32 + mf * 16 + (long)hi * 8;
#pragma unroll
        for (int r = 0; r < 8; ++r)
          T[(rowb + r) * RPAD + col] = (__bf16)acc[mf][nf][r];
      }
    }
}

// ---------------- kernel 3: fused gate/up + SiLU --------------------------
// h = silu(x@w1^T + t1@u1^T) * (x@w3^T + t3@u3^T), stored bf16 [8192, 7168]

__global__ void __launch_bounds__(256, 2)
k_gateup(const float* __restrict__ X, const float* __restrict__ W1,
         const float* __restrict__ W3, const __bf16* __restrict__ T1,
         const __bf16* __restrict__ T3, const float* __restrict__ U1,
         const float* __restrict__ U3, __bf16* __restrict__ H) {
  __shared__ __bf16 sm0[TLE];   // A tile
  __shared__ __bf16 sm1[TLE];   // B tile (gate path)
  __shared__ __bf16 sm2[TLE];   // B tile (up path)
  const int tid = threadIdx.x;
  const long m0 = (long)blockIdx.x * BM;
  const long i0 = (long)blockIdx.y * BN;
  const int lane = tid & 31, wv = tid >> 5;
  const int wm = wv & 3, wn = wv >> 2;
  const int lane16 = lane & 15, hi = lane >> 4;
  const v8f z = {0.f, 0.f, 0.f, 0.f, 0.f, 0.f, 0.f, 0.f};
  v8f accg[2][4], accu[2][4];
  for (int a = 0; a < 2; ++a)
    for (int b = 0; b < 4; ++b) { accg[a][b] = z; accu[a][b] = z; }

  // dense phase over H=2048 (shared A tile, two B tiles)
  for (long k0 = 0; k0 < HDIM; k0 += BK) {
    ld_tile_f32v4(sm0, X,  m0, k0, HDIM, BIGROW, tid);
    ld_tile_f32v4(sm1, W1, i0, k0, HDIM, BIGROW, tid);
    ld_tile_f32v4(sm2, W3, i0, k0, HDIM, BIGROW, tid);
    __syncthreads();
    mma_step(sm0, sm1, accg, wm, wn, lane16, hi);
    mma_step(sm0, sm2, accu, wm, wn, lane16, hi);
    __syncthreads();
  }
  // low-rank gate: t1 @ u1^T  (K padded to 448; u1 K>=398 zero-filled)
  for (long k0 = 0; k0 < RPAD; k0 += BK) {
    ld_tile_bf16(sm0, T1, m0, k0, RPAD, tid);
    ld_tile_f32s(sm1, U1, i0, k0, RREAL, RREAL, tid);
    tile_barrier();
    mma_step(sm0, sm1, accg, wm, wn, lane16, hi);
    __syncthreads();
  }
  // low-rank up: t3 @ u3^T
  for (long k0 = 0; k0 < RPAD; k0 += BK) {
    ld_tile_bf16(sm0, T3, m0, k0, RPAD, tid);
    ld_tile_f32s(sm1, U3, i0, k0, RREAL, RREAL, tid);
    tile_barrier();
    mma_step(sm0, sm1, accu, wm, wn, lane16, hi);
    __syncthreads();
  }
  // epilogue: silu(gate) * up -> bf16
#pragma unroll
  for (int mf = 0; mf < 2; ++mf)
#pragma unroll
    for (int nf = 0; nf < 4; ++nf) {
      long col  = i0 + wn * 64 + nf * 16 + lane16;
      long rowb = m0 + wm * 32 + mf * 16 + (long)hi * 8;
#pragma unroll
      for (int r = 0; r < 8; ++r) {
        float g = accg[mf][nf][r];
        float u = accu[mf][nf][r];
        float h = (g / (1.f + __expf(-g))) * u;
        H[(rowb + r) * IDIM + col] = (__bf16)h;
      }
    }
}

// ---------------- kernel 6: down projection + scale -----------------------
// out = (h@w2^T + t2@u2^T) * scale[token], f32 output

__global__ void __launch_bounds__(256, 2)
k_out(const __bf16* __restrict__ Hb, const float* __restrict__ W2,
      const __bf16* __restrict__ T2, const float* __restrict__ U2,
      const float* __restrict__ scale, float* __restrict__ Out) {
  __shared__ __bf16 sm0[TLE];
  __shared__ __bf16 sm1[TLE];
  const int tid = threadIdx.x;
  const long m0 = (long)blockIdx.x * BM;
  const long n0 = (long)blockIdx.y * BN;
  const int lane = tid & 31, wv = tid >> 5;
  const int wm = wv & 3, wn = wv >> 2;
  const int lane16 = lane & 15, hi = lane >> 4;
  const v8f z = {0.f, 0.f, 0.f, 0.f, 0.f, 0.f, 0.f, 0.f};
  v8f acc[2][4];
  for (int a = 0; a < 2; ++a)
    for (int b = 0; b < 4; ++b) acc[a][b] = z;

  for (long k0 = 0; k0 < IDIM; k0 += BK) {
    ld_tile_bf16(sm0, Hb, m0, k0, IDIM, tid);
    ld_tile_f32v4(sm1, W2, n0, k0, IDIM, BIGROW, tid);
    tile_barrier();
    mma_step(sm0, sm1, acc, wm, wn, lane16, hi);
    __syncthreads();
  }
  for (long k0 = 0; k0 < RPAD; k0 += BK) {
    ld_tile_bf16(sm0, T2, m0, k0, RPAD, tid);
    ld_tile_f32s(sm1, U2, n0, k0, RREAL, RREAL, tid);
    tile_barrier();
    mma_step(sm0, sm1, acc, wm, wn, lane16, hi);
    __syncthreads();
  }
#pragma unroll
  for (int mf = 0; mf < 2; ++mf)
#pragma unroll
    for (int nf = 0; nf < 4; ++nf) {
      long col  = n0 + wn * 64 + nf * 16 + lane16;
      long rowb = m0 + wm * 32 + mf * 16 + (long)hi * 8;
#pragma unroll
      for (int r = 0; r < 8; ++r) {
        long row = rowb + r;
        Out[row * HDIM + col] = acc[mf][nf][r] * scale[row];
      }
    }
}

// ---------------- host launch ---------------------------------------------

extern "C" void kernel_launch(void* const* d_in, const int* in_sizes, int n_in,
                              void* d_out, int out_size, void* d_ws,
                              size_t ws_size, hipStream_t stream) {
  (void)in_sizes; (void)n_in; (void)out_size; (void)ws_size;
  const float* x  = (const float*)d_in[0];
  const float* gw = (const float*)d_in[1];
  const float* w1 = (const float*)d_in[2];
  const float* w2 = (const float*)d_in[3];
  const float* w3 = (const float*)d_in[4];
  const float* u1 = (const float*)d_in[5];
  const float* v1 = (const float*)d_in[6];
  const float* u2 = (const float*)d_in[7];
  const float* v2 = (const float*)d_in[8];
  const float* u3 = (const float*)d_in[9];
  const float* v3 = (const float*)d_in[10];

  float* out    = (float*)d_out;
  float* logits = out + NTOK * HDIM;    // second return value, concatenated

  char*   ws    = (char*)d_ws;
  float*  scale = (float*)ws;                      // 8192 f32
  __bf16* t1    = (__bf16*)(ws + 65536);
  __bf16* t3    = t1 + NTOK * RPAD;
  __bf16* t2    = t3 + NTOK * RPAD;
  __bf16* hb    = t2 + NTOK * RPAD;                // 8192 x 7168 bf16

  const dim3 blk(256);
  k_router<<<NTOK, 256, 0, stream>>>(x, gw, logits, scale);
  k_lowrank<float><<<dim3(64, 4), blk, 0, stream>>>(x, v1, t1, HDIM);
  k_lowrank<float><<<dim3(64, 4), blk, 0, stream>>>(x, v3, t3, HDIM);
  k_gateup<<<dim3(64, 56), blk, 0, stream>>>(x, w1, w3, t1, t3, u1, u3, hb);
  k_lowrank<__bf16><<<dim3(64, 4), blk, 0, stream>>>(hb, v2, t2, IDIM);
  k_out<<<dim3(64, 16), blk, 0, stream>>>(hb, w2, t2, u2, scale, out);
}
